// MultiHeadAttention_19576460935441
// MI455X (gfx1250) — compile-verified, matrix operand order checked by reading, not run on hardware
//
#include <hip/hip_runtime.h>

// ---- problem constants (match reference) ----
#define HH 8
#define DM 512
#define HD 64
#define NN 4096
#define QSCALE 0.125f   // HEAD_DIM^-0.5, 64^-0.5
#define NEGBIG (-3.0e38f)

// ---- CDNA5 WMMA types ----
typedef __attribute__((ext_vector_type(16))) __bf16        v16bf;
typedef __attribute__((ext_vector_type(8)))  float         v8f;
typedef __attribute__((ext_vector_type(4)))  float         v4f;
typedef __attribute__((ext_vector_type(4)))  unsigned int  v4u;

union ABregs {                 // one WMMA 16-bit A/B operand (8 VGPRs)
    v16bf          v;
    unsigned short s[16];
    v4u            q[2];
};

__device__ __forceinline__ unsigned short f2bf(float f) {
    unsigned int u = __float_as_uint(f);
    u += 0x7FFFu + ((u >> 16) & 1u);      // round-to-nearest-even
    return (unsigned short)(u >> 16);
}

__device__ __forceinline__ void pack8(ABregs& r, int base, v4f lo, v4f hi) {
#pragma unroll
    for (int i = 0; i < 4; ++i) {
        r.s[base + i]     = f2bf(lo[i]);
        r.s[base + 4 + i] = f2bf(hi[i]);
    }
}

// A-operand load, fp32 source (convert to bf16). p = row base, k = kt + khalf.
// Layout (ISA 7.12.2, 16-bit A 16x32): lane<16 -> K {0..7,16..23}, lane>=16 -> K {8..15,24..31}
__device__ __forceinline__ void loadA(ABregs& a, const float* __restrict__ p, int k) {
    v4f x0 = *(const v4f*)(p + k);
    v4f x1 = *(const v4f*)(p + k + 4);
    v4f x2 = *(const v4f*)(p + k + 16);
    v4f x3 = *(const v4f*)(p + k + 20);
    pack8(a, 0, x0, x1);
    pack8(a, 8, x2, x3);
}
// A-operand load, bf16 source (raw b128 loads)
__device__ __forceinline__ void loadA(ABregs& a, const unsigned short* __restrict__ p, int k) {
    a.q[0] = *(const v4u*)(p + k);
    a.q[1] = *(const v4u*)(p + k + 16);
}

__device__ __forceinline__ v8f wmma_bf16(const ABregs& a, const ABregs& b, v8f c) {
    return __builtin_amdgcn_wmma_f32_16x16x32_bf16(
        /*neg_a=*/false, a.v, /*neg_b=*/false, b.v,
        /*c_mod=*/(short)0, c, /*reuse_a=*/false, /*reuse_b=*/false);
}

// =====================================================================
// GEMM:  Y[M,N] = A[M,DM] @ W[N,DM]^T + bias, with flexible output layout
//   mode 0: Y -> bf16 [H][NN][64]   (Q, K; scale folded in)
//   mode 1: Y -> bf16 [H][64][NN]   (V transposed, for contiguous PV loads)
//   mode 2: Y -> fp32 [M][DM]       (final output)
// Block = 256 threads (8 waves); each block owns 16 rows x 512 cols,
// wave w owns cols [64w, 64w+64) as 4 WMMA subtiles. K loop: 16 x (K=32).
// =====================================================================
template <typename TA>
__global__ __launch_bounds__(256) void gemm_wt_bias(
    const TA* __restrict__ A, const float* __restrict__ W,
    const float* __restrict__ bias, void* __restrict__ out,
    int mode, float scale)
{
    const int lane  = threadIdx.x & 31;
    const int wave  = threadIdx.x >> 5;
    const int mtile = blockIdx.x * 16;
    const int ln    = lane & 15;
    const int hi    = lane >> 4;
    const int khalf = hi * 8;
    const int mrow  = mtile + ln;          // A row this lane supplies
    const int cbase = wave * 64;

    v8f acc[4] = {};
    for (int kt = 0; kt < DM; kt += 32) {
        ABregs a;
        loadA(a, A + (size_t)mrow * DM, kt + khalf);
#pragma unroll
        for (int sub = 0; sub < 4; ++sub) {
            const int col = cbase + sub * 16 + ln;           // B column = lane&15
            const float* wp = W + (size_t)col * DM + kt + khalf;
            ABregs b;                                         // B[k,n] = W[n,k] (contiguous in k)
            v4f w0 = *(const v4f*)(wp);
            v4f w1 = *(const v4f*)(wp + 4);
            v4f w2 = *(const v4f*)(wp + 16);
            v4f w3 = *(const v4f*)(wp + 20);
            pack8(b, 0, w0, w1);
            pack8(b, 8, w2, w3);
            acc[sub] = wmma_bf16(a, b, acc[sub]);
        }
    }
    // epilogue: C layout -> lane holds (M = r + 8*hi, N = ln)
#pragma unroll
    for (int sub = 0; sub < 4; ++sub) {
        const int col = cbase + sub * 16 + ln;
        const float bv = bias[col];
#pragma unroll
        for (int r = 0; r < 8; ++r) {
            const int row = mtile + r + hi * 8;
            const float y = (acc[sub][r] + bv) * scale;
            if (mode == 2) {
                ((float*)out)[(size_t)row * DM + col] = y;
            } else {
                const unsigned short bf = f2bf(y);
                const int h = col >> 6, d = col & 63;
                if (mode == 0)
                    ((unsigned short*)out)[((size_t)h * NN + row) * HD + d] = bf;
                else
                    ((unsigned short*)out)[((size_t)h * HD + d) * NN + row] = bf;
            }
        }
    }
}

// =====================================================================
// Flash-style attention. Block = 256 threads; wave w handles head w for
// the 16-row tile blockIdx.x. Per step: 64 keys -> 8 WMMA (S) + 8 WMMA (PV).
// Online softmax in fp32 (ds_bpermute butterflies amortized over 64 keys),
// batch-mask with finite -3e38 sentinel (self-correcting, no NaNs).
// Bias streamed with non-temporal loads (512MB, zero reuse) to keep the
// 192MB L2 for K/V/Q (12MB, reused 256x). P transposed C-layout -> A-layout
// via a 2KB/wave LDS round trip.
// =====================================================================
__global__ __launch_bounds__(256) void attn_kernel(
    const unsigned short* __restrict__ Qb,   // bf16 [H][NN][64], pre-scaled
    const unsigned short* __restrict__ Kb,   // bf16 [H][NN][64]
    const unsigned short* __restrict__ Vt,   // bf16 [H][64][NN]
    const float* __restrict__ bias,          // fp32 [H][NN][NN]
    const int* __restrict__ batch,           // int32 [NN]
    unsigned short* __restrict__ AO)         // bf16 [NN][DM]
{
    __shared__ unsigned short plds[HH][16][64];

    const int lane  = threadIdx.x & 31;
    const int h     = threadIdx.x >> 5;      // 8 waves == 8 heads
    const int i0    = blockIdx.x * 16;
    const int ln    = lane & 15;
    const int hi    = lane >> 4;
    const int khalf = hi * 8;

    // Q tile, A layout, split into K-chunks [0,32) and [32,64)
    const unsigned short* qp = Qb + ((size_t)h * NN + i0 + ln) * HD;
    ABregs qa0, qa1;
    qa0.q[0] = *(const v4u*)(qp + khalf);
    qa0.q[1] = *(const v4u*)(qp + khalf + 16);
    qa1.q[0] = *(const v4u*)(qp + 32 + khalf);
    qa1.q[1] = *(const v4u*)(qp + 32 + khalf + 16);

    int rb[8];                                // batch id of this lane's 8 rows
#pragma unroll
    for (int r = 0; r < 8; ++r) rb[r] = batch[i0 + r + hi * 8];

    const float* bp = bias + ((size_t)h * NN + i0) * NN;

    float m[8], l[8];
    v8f o[4] = {};
#pragma unroll
    for (int r = 0; r < 8; ++r) { m[r] = NEGBIG; l[r] = 0.f; }

    for (int jb = 0; jb < NN; jb += 64) {
        // ---- S = Q K^T over 64 keys (4 x 16x16 fp32 accumulators) ----
        v8f s[4] = {};
#pragma unroll
        for (int g = 0; g < 4; ++g) {
            const unsigned short* kp = Kb + ((size_t)h * NN + jb + g * 16 + ln) * HD;
            ABregs b;
            b.q[0] = *(const v4u*)(kp + khalf);
            b.q[1] = *(const v4u*)(kp + khalf + 16);
            s[g] = wmma_bf16(qa0, b, s[g]);
            b.q[0] = *(const v4u*)(kp + 32 + khalf);
            b.q[1] = *(const v4u*)(kp + 32 + khalf + 16);
            s[g] = wmma_bf16(qa1, b, s[g]);
        }

        // ---- bias + batch mask + online softmax (fp32) ----
        int cb[4];
#pragma unroll
        for (int g = 0; g < 4; ++g) cb[g] = batch[jb + g * 16 + ln];

        float p[4][8], fac[8];
#pragma unroll
        for (int r = 0; r < 8; ++r) {
            const float* br = bp + (size_t)(r + hi * 8) * NN + jb;
            float v[4];
#pragma unroll
            for (int g = 0; g < 4; ++g) {
                const float t = s[g][r] + __builtin_nontemporal_load(br + g * 16 + ln);
                v[g] = (cb[g] == rb[r]) ? t : NEGBIG;
            }
            float mx = fmaxf(fmaxf(v[0], v[1]), fmaxf(v[2], v[3]));
            mx = fmaxf(mx, __shfl_xor(mx, 1, 32));
            mx = fmaxf(mx, __shfl_xor(mx, 2, 32));
            mx = fmaxf(mx, __shfl_xor(mx, 4, 32));
            mx = fmaxf(mx, __shfl_xor(mx, 8, 32));
            const float mn = fmaxf(m[r], mx);
            float sum = 0.f;
#pragma unroll
            for (int g = 0; g < 4; ++g) {
                const float e = __expf(v[g] - mn);
                p[g][r] = e;
                sum += e;
            }
            sum += __shfl_xor(sum, 1, 32);
            sum += __shfl_xor(sum, 2, 32);
            sum += __shfl_xor(sum, 4, 32);
            sum += __shfl_xor(sum, 8, 32);
            const float f = __expf(m[r] - mn);
            l[r] = l[r] * f + sum;
            m[r] = mn;
            fac[r] = f;
        }
        // rescale running output
#pragma unroll
        for (int d = 0; d < 4; ++d)
#pragma unroll
            for (int r = 0; r < 8; ++r) o[d][r] = o[d][r] * fac[r];

        // ---- P: C layout -> A layout via LDS ----
        __syncthreads();
#pragma unroll
        for (int r = 0; r < 8; ++r)
#pragma unroll
            for (int g = 0; g < 4; ++g)
                plds[h][r + hi * 8][g * 16 + ln] = f2bf(p[g][r]);
        __syncthreads();
        ABregs pa0, pa1;
        pa0.q[0] = *(const v4u*)&plds[h][ln][khalf];
        pa0.q[1] = *(const v4u*)&plds[h][ln][khalf + 16];
        pa1.q[0] = *(const v4u*)&plds[h][ln][32 + khalf];
        pa1.q[1] = *(const v4u*)&plds[h][ln][32 + khalf + 16];

        // ---- O += P @ V (4 x 16x16 output subtiles over HEAD_DIM, K=64) ----
#pragma unroll
        for (int d = 0; d < 4; ++d) {
            const unsigned short* vp =
                Vt + ((size_t)h * HD + d * 16 + ln) * NN + jb;   // B[k,n]=Vt[n_dim][key]
            ABregs b;
            b.q[0] = *(const v4u*)(vp + khalf);
            b.q[1] = *(const v4u*)(vp + khalf + 16);
            o[d] = wmma_bf16(pa0, b, o[d]);
            b.q[0] = *(const v4u*)(vp + 32 + khalf);
            b.q[1] = *(const v4u*)(vp + 32 + khalf + 16);
            o[d] = wmma_bf16(pa1, b, o[d]);
        }
    }

    // ---- normalize and store bf16 attention output [NN][DM] ----
    float inv[8];
#pragma unroll
    for (int r = 0; r < 8; ++r) inv[r] = (l[r] > 0.f) ? 1.0f / l[r] : 0.f;
#pragma unroll
    for (int d = 0; d < 4; ++d) {
        const int col = h * HD + d * 16 + ln;
#pragma unroll
        for (int r = 0; r < 8; ++r) {
            const int row = i0 + r + hi * 8;
            AO[(size_t)row * DM + col] = f2bf(o[d][r] * inv[r]);
        }
    }
}

// =====================================================================
extern "C" void kernel_launch(void* const* d_in, const int* in_sizes, int n_in,
                              void* d_out, int out_size, void* d_ws, size_t ws_size,
                              hipStream_t stream)
{
    (void)in_sizes; (void)n_in; (void)out_size; (void)ws_size;
    const float* x     = (const float*)d_in[0];
    const float* abias = (const float*)d_in[1];
    const int*   batch = (const int*)d_in[2];
    const float* Wq = (const float*)d_in[3];  const float* bq = (const float*)d_in[4];
    const float* Wk = (const float*)d_in[5];  const float* bk = (const float*)d_in[6];
    const float* Wv = (const float*)d_in[7];  const float* bv = (const float*)d_in[8];
    const float* Wo = (const float*)d_in[9];  const float* bo = (const float*)d_in[10];
    float* out = (float*)d_out;

    // workspace: Qb | Kb | Vt | AO  (each H*N*64 = N*DM bf16 = 4 MB)
    unsigned short* Qb = (unsigned short*)d_ws;
    unsigned short* Kb = Qb + (size_t)HH * NN * HD;
    unsigned short* Vt = Kb + (size_t)HH * NN * HD;
    unsigned short* AO = Vt + (size_t)HH * NN * HD;

    const dim3 blk(256);
    const dim3 grd(NN / 16);
    gemm_wt_bias<float><<<grd, blk, 0, stream>>>(x, Wq, bq, Qb, 0, QSCALE);
    gemm_wt_bias<float><<<grd, blk, 0, stream>>>(x, Wk, bk, Kb, 0, 1.0f);
    gemm_wt_bias<float><<<grd, blk, 0, stream>>>(x, Wv, bv, Vt, 1, 1.0f);
    attn_kernel<<<dim3(NN / 16), dim3(256), 0, stream>>>(Qb, Kb, Vt, abias, batch, AO);
    gemm_wt_bias<unsigned short><<<grd, blk, 0, stream>>>(AO, Wo, bo, out, 2, 1.0f);
}
